// UniMolLayer_65317862637739
// MI455X (gfx1250) — matrix operand order repaired and checked by reading, not compiled, not run on hardware
//
#include <hip/hip_runtime.h>
#include <hip/hip_bf16.h>
#include <cstdint>

#define B_ 32
#define N_ 512
#define E_ 512
#define H_ 16
#define F_ 2048
#define M_ (B_ * N_)   // 16384 rows

typedef __attribute__((ext_vector_type(16))) _Float16 v16h;
typedef __attribute__((ext_vector_type(8)))  _Float16 v8h;
typedef __attribute__((ext_vector_type(8)))  float    v8f;

// ---------------------------------------------------------------------------
// WMMA helpers (CDNA5 wave32, V_WMMA_F32_16X16X32_F16)
// ---------------------------------------------------------------------------
__device__ __forceinline__ v8f wmma16x16x32(v16h a, v16h b, v8f c) {
  // (neg_a, A, neg_b, B, c_mod, C, reuse_a, reuse_b)
  return __builtin_amdgcn_wmma_f32_16x16x32_f16(false, a, false, b, (short)0, c,
                                                false, false);
}

// A fragment: 16x32 (MxK) f16, row-major source with leading dim lda.
// Lane layout: lanes 0-15 M=lane, K=0..15;
//              lanes 16-31 M=lane-16, K=16..31.
__device__ __forceinline__ v16h load_a_frag(const _Float16* A, int lda) {
  int lane = threadIdx.x & 31;
  const _Float16* p = A + (size_t)(lane & 15) * lda + ((lane >> 4) << 4);
  v8h lo = *(const v8h*)(p);        // K = kb .. kb+7
  v8h hi = *(const v8h*)(p + 8);    // K = kb+8 .. kb+15
  v16h r;
#pragma unroll
  for (int i = 0; i < 8; ++i) { r[i] = lo[i]; r[i + 8] = hi[i]; }
  return r;
}

// B fragment: 32x16 (KxN) f16, from a TRANSPOSED (N x K) row-major source BT.
// Lanes 0-15 hold N=lane with K=0..15; lanes 16-31 hold N=lane-16, K=16..31.
__device__ __forceinline__ v16h load_b_frag(const _Float16* BT, int ldb) {
  int lane = threadIdx.x & 31;
  const _Float16* p = BT + (size_t)(lane & 15) * ldb + ((lane >> 4) << 4);
  v8h lo = *(const v8h*)(p);
  v8h hi = *(const v8h*)(p + 8);
  v16h r;
#pragma unroll
  for (int i = 0; i < 8; ++i) { r[i] = lo[i]; r[i + 8] = hi[i]; }
  return r;
}

// B fragment from a row-major K x N source (strided per-lane gather).
__device__ __forceinline__ v16h load_b_frag_strided(const _Float16* Brm, int ldb) {
  int lane = threadIdx.x & 31;
  int n = lane & 15;
  int kb = (lane >> 4) << 4;
  v16h r;
#pragma unroll
  for (int j = 0; j < 16; ++j) r[j] = Brm[(size_t)(kb + j) * ldb + n];
  return r;
}

// ---------------------------------------------------------------------------
// CDNA5 async global->LDS copy (ASYNCcnt-tracked Tensor/async path)
// ---------------------------------------------------------------------------
__device__ __forceinline__ void async_load_b128_to_lds(uint32_t lds_off,
                                                       const void* gaddr) {
  asm volatile("global_load_async_to_lds_b128 %0, %1, off"
               :: "v"(lds_off), "v"(gaddr) : "memory");
}
__device__ __forceinline__ void wait_asynccnt0() {
  asm volatile("s_wait_asynccnt 0x0" ::: "memory");
}
__device__ __forceinline__ void wait_asynccnt4() {
  asm volatile("s_wait_asynccnt 0x4" ::: "memory");
}

// ---------------------------------------------------------------------------
// Fused WMMA GEMM:  out = epilogue(A(f16, MxK) * BT(f16, NxK)^T + bias)
// 128 threads = 4 waves/block; each wave owns a 32x64 C tile
// (2 A-frags x 4 B-frags -> 8 WMMAs per 32-deep K chunk).
// ---------------------------------------------------------------------------
template <bool GELU, bool RESID, bool OUT32, bool OUT16>
__global__ __launch_bounds__(128) void gemm_wmma(
    const _Float16* __restrict__ A, const _Float16* __restrict__ BT,
    const float* __restrict__ bias, const float* __restrict__ resid,
    float* __restrict__ outF, _Float16* __restrict__ outH,
    int M, int Nc, int K) {
  int wave = threadIdx.x >> 5;
  int n0 = (blockIdx.x * 4 + wave) * 64;
  int m0 = blockIdx.y * 32;
  const _Float16* Ab = A + (size_t)m0 * K;
  const _Float16* Bb = BT + (size_t)n0 * K;
  const v8f vzero = {};
  v8f acc[2][4];
#pragma unroll
  for (int i = 0; i < 2; ++i)
#pragma unroll
    for (int j = 0; j < 4; ++j) acc[i][j] = vzero;

  for (int k = 0; k < K; k += 32) {
    if (k + 32 < K) {
      __builtin_prefetch((const void*)(Ab + k + 32), 0, 1);
      __builtin_prefetch((const void*)(Bb + k + 32), 0, 1);
    }
    v16h a0 = load_a_frag(Ab + k, K);
    v16h a1 = load_a_frag(Ab + (size_t)16 * K + k, K);
    v16h b0 = load_b_frag(Bb + k, K);
    v16h b1 = load_b_frag(Bb + (size_t)16 * K + k, K);
    v16h b2 = load_b_frag(Bb + (size_t)32 * K + k, K);
    v16h b3 = load_b_frag(Bb + (size_t)48 * K + k, K);
    acc[0][0] = wmma16x16x32(a0, b0, acc[0][0]);
    acc[0][1] = wmma16x16x32(a0, b1, acc[0][1]);
    acc[0][2] = wmma16x16x32(a0, b2, acc[0][2]);
    acc[0][3] = wmma16x16x32(a0, b3, acc[0][3]);
    acc[1][0] = wmma16x16x32(a1, b0, acc[1][0]);
    acc[1][1] = wmma16x16x32(a1, b1, acc[1][1]);
    acc[1][2] = wmma16x16x32(a1, b2, acc[1][2]);
    acc[1][3] = wmma16x16x32(a1, b3, acc[1][3]);
  }

  // C layout: VGPR r, lane l -> row r + 8*(l>=16), col l&15
  int lane = threadIdx.x & 31;
  int colL = lane & 15;
  int rhalf = (lane >> 4) << 3;
#pragma unroll
  for (int j = 0; j < 4; ++j) {
    int col = n0 + j * 16 + colL;
    float bv = bias[col];
#pragma unroll
    for (int i = 0; i < 2; ++i) {
#pragma unroll
      for (int r = 0; r < 8; ++r) {
        int row = m0 + i * 16 + rhalf + r;
        float v = acc[i][j][r] + bv;
        if (GELU) v = 0.5f * v * (1.0f + erff(v * 0.70710678118654752f));
        if (RESID) v += resid[(size_t)row * Nc + col];
        if (OUT32) outF[(size_t)row * Nc + col] = v;
        if (OUT16) outH[(size_t)row * Nc + col] = (_Float16)v;
      }
    }
  }
}

// ---------------------------------------------------------------------------
// LayerNorm (one 256-thread block per row of 512), f32 in -> f16 out
// ---------------------------------------------------------------------------
__global__ __launch_bounds__(256) void layernorm_f16(
    const float* __restrict__ x, const float* __restrict__ g,
    const float* __restrict__ bta, _Float16* __restrict__ out) {
  int row = blockIdx.x;
  const float* xr = x + (size_t)row * E_;
  int t = threadIdx.x;
  float v0 = xr[t], v1 = xr[t + 256];
  float s = v0 + v1, s2 = v0 * v0 + v1 * v1;
#pragma unroll
  for (int off = 16; off > 0; off >>= 1) {
    s += __shfl_xor(s, off);
    s2 += __shfl_xor(s2, off);
  }
  __shared__ float ls[8], ls2[8];
  if ((t & 31) == 0) { ls[t >> 5] = s; ls2[t >> 5] = s2; }
  __syncthreads();
  float S = 0.f, S2 = 0.f;
#pragma unroll
  for (int i = 0; i < 8; ++i) { S += ls[i]; S2 += ls2[i]; }
  float mu = S * (1.0f / E_);
  float var = S2 * (1.0f / E_) - mu * mu;
  float rstd = rsqrtf(var + 1e-5f);
  _Float16* orow = out + (size_t)row * E_;
  orow[t]       = (_Float16)((v0 - mu) * rstd * g[t] + bta[t]);
  orow[t + 256] = (_Float16)((v1 - mu) * rstd * g[t + 256] + bta[t + 256]);
}

// ---------------------------------------------------------------------------
// Attention: one wave per (b, h, 16-query tile).
// scores = Q K^T / sqrt(D) + pair*pw + pb (+row mask) -> softmax -> P V
// V tiles are staged global->LDS with async copies (double-buffered).
// ---------------------------------------------------------------------------
__global__ __launch_bounds__(32) void attn_kernel(
    const _Float16* __restrict__ qkv, const float* __restrict__ pair,
    const unsigned char* __restrict__ mask, const float* __restrict__ pair_w,
    const float* __restrict__ pair_b, _Float16* __restrict__ attn_out) {
  __shared__ float    s_scores[16 * N_];     // 32 KB
  __shared__ _Float16 s_probs[16 * N_];      // 16 KB
  __shared__ _Float16 s_vtile[2][32 * 32];   // 4 KB, double-buffered V tile
  __shared__ float    s_stat[48];            // rowmax | 1/rowsum | maskbias

  int nt = blockIdx.x, h = blockIdx.y, b = blockIdx.z;
  int n0 = nt * 16;
  int lane = threadIdx.x;
  int col = lane & 15;
  int rhalf = (lane >> 4) << 3;
  const float pw = pair_w[h], pb = pair_b[h];
  const float rscale = 0.17677669529663687f;  // 1/sqrt(32)

  if (lane < 16)
    s_stat[32 + lane] = mask[(size_t)b * N_ + n0 + lane] ? 0.f : -1e9f;
  __syncthreads();

  // Q fragment (16 x 32), qkv layout: (B,N,3,H,D) -> stride 1536 per row
  const _Float16* qbase = qkv + ((size_t)(b * N_ + n0)) * 1536 + h * 32;
  v16h qa = load_a_frag(qbase, 1536);

  // Pass 1: scores into LDS
  for (int m0c = 0; m0c < N_; m0c += 16) {
    const _Float16* kbase =
        qkv + ((size_t)(b * N_ + m0c)) * 1536 + E_ + h * 32;
    v16h kb = load_b_frag(kbase, 1536);  // K tile is row-major (m,d) == B^T
    v8f c = {};
    c = wmma16x16x32(qa, kb, c);
#pragma unroll
    for (int r = 0; r < 8; ++r) {
      int row = rhalf + r;
      float pr = pair[((size_t)b * N_ + (n0 + row)) * N_ + m0c + col];
      s_scores[row * N_ + m0c + col] =
          c[r] * rscale + pr * pw + pb + s_stat[32 + row];
    }
  }
  __syncthreads();

  // Pass 2: row max and sum (each row handled by a lane pair)
  {
    int row = lane & 15, half = lane >> 4;
    const float* sr = s_scores + row * N_ + half * 256;
    float mx = -3.4e38f;
    for (int i = 0; i < 256; ++i) mx = fmaxf(mx, sr[i]);
    mx = fmaxf(mx, __shfl_xor(mx, 16));
    float sum = 0.f;
    for (int i = 0; i < 256; ++i) sum += __expf(sr[i] - mx);
    sum += __shfl_xor(sum, 16);
    if (lane < 16) { s_stat[lane] = mx; s_stat[16 + lane] = 1.0f / sum; }
  }
  __syncthreads();

  // Pass 3: f16 probabilities
  for (int idx = lane; idx < 16 * N_; idx += 32) {
    int row = idx >> 9;  // N_ == 512
    float p = __expf(s_scores[idx] - s_stat[row]) * s_stat[16 + row];
    s_probs[idx] = (_Float16)p;
  }
  __syncthreads();

  // Pass 4: P @ V. Each lane async-copies one V row (32 f16 = 4 x B128).
  auto issueV = [&](int mc, int buf) {
    const _Float16* vrow =
        qkv + ((size_t)(b * N_ + mc + lane)) * 1536 + 2 * E_ + h * 32;
    uint32_t ldsrow = (uint32_t)(uintptr_t)(&s_vtile[buf][lane * 32]);
#pragma unroll
    for (int seg = 0; seg < 4; ++seg)
      async_load_b128_to_lds(ldsrow + seg * 16, (const void*)(vrow + seg * 8));
  };

  v8f c0 = {}, c1 = {};
  issueV(0, 0);
  int buf = 0;
  for (int mc = 0; mc < N_; mc += 32) {
    if (mc + 32 < N_) {
      issueV(mc + 32, buf ^ 1);
      wait_asynccnt4();  // first group (4 async ops) retired, in-order
    } else {
      wait_asynccnt0();
    }
    v16h pa = load_a_frag(s_probs + mc, N_);
    const _Float16* vb = &s_vtile[buf][0];
    v16h b0 = load_b_frag_strided(vb, 32);       // d = 0..15
    v16h b1 = load_b_frag_strided(vb + 16, 32);  // d = 16..31
    c0 = wmma16x16x32(pa, b0, c0);
    c1 = wmma16x16x32(pa, b1, c1);
    buf ^= 1;
  }
  _Float16* obase = attn_out + ((size_t)(b * N_ + n0)) * E_ + h * 32;
#pragma unroll
  for (int r = 0; r < 8; ++r) {
    int row = rhalf + r;
    obase[(size_t)row * E_ + col]      = (_Float16)c0[r];
    obase[(size_t)row * E_ + 16 + col] = (_Float16)c1[r];
  }
}

// ---------------------------------------------------------------------------
// Utility kernels
// ---------------------------------------------------------------------------
// src: Krows x Ncols f32 (row-major) -> dst: Ncols x Krows f16 (row-major)
__global__ void transpose_f32_to_f16(const float* __restrict__ src,
                                     _Float16* __restrict__ dst, int Krows,
                                     int Ncols) {
  size_t i = (size_t)blockIdx.x * blockDim.x + threadIdx.x;
  if (i >= (size_t)Krows * Ncols) return;
  int n = (int)(i / Krows);
  int k = (int)(i % Krows);
  dst[i] = (_Float16)src[(size_t)k * Ncols + n];
}

__global__ void copy_f32x4(const float4* __restrict__ s,
                           float4* __restrict__ d) {
  size_t i = (size_t)blockIdx.x * blockDim.x + threadIdx.x;
  d[i] = s[i];
}

// ---------------------------------------------------------------------------
// Host launcher
// ---------------------------------------------------------------------------
extern "C" void kernel_launch(void* const* d_in, const int* in_sizes, int n_in,
                              void* d_out, int out_size, void* d_ws,
                              size_t ws_size, hipStream_t stream) {
  const float* atom_embed = (const float*)d_in[0];
  const float* pair_repr  = (const float*)d_in[1];
  const unsigned char* mask = (const unsigned char*)d_in[2];
  const float* qkv_w = (const float*)d_in[3];
  const float* qkv_b = (const float*)d_in[4];
  const float* out_w = (const float*)d_in[5];
  const float* out_b = (const float*)d_in[6];
  const float* pair_w = (const float*)d_in[7];
  const float* pair_b = (const float*)d_in[8];
  const float* ln1_g = (const float*)d_in[9];
  const float* ln1_b = (const float*)d_in[10];
  const float* ln2_g = (const float*)d_in[11];
  const float* ln2_b = (const float*)d_in[12];
  const float* w1 = (const float*)d_in[13];
  const float* b1 = (const float*)d_in[14];
  const float* w2 = (const float*)d_in[15];
  const float* b2 = (const float*)d_in[16];
  float* out = (float*)d_out;

  // Workspace carve-up (all sizes 256B-aligned)
  char* ws = (char*)d_ws;
  _Float16* x16    = (_Float16*)ws;  ws += (size_t)M_ * E_ * 2;        // 16 MB
  _Float16* qkv16  = (_Float16*)ws;  ws += (size_t)M_ * 3 * E_ * 2;    // 48 MB
  _Float16* attn16 = (_Float16*)ws;  ws += (size_t)M_ * E_ * 2;        // 16 MB
  _Float16* y16    = (_Float16*)ws;  ws += (size_t)M_ * E_ * 2;        // 16 MB
  _Float16* h16    = (_Float16*)ws;  ws += (size_t)M_ * F_ * 2;        // 64 MB
  _Float16* qkvwT  = (_Float16*)ws;  ws += (size_t)E_ * 3 * E_ * 2;
  _Float16* outwT  = (_Float16*)ws;  ws += (size_t)E_ * E_ * 2;
  _Float16* w1T    = (_Float16*)ws;  ws += (size_t)E_ * F_ * 2;
  _Float16* w2T    = (_Float16*)ws;  ws += (size_t)F_ * E_ * 2;

  // 1) weight conversion f32 -> f16 transposed (N x K)
  transpose_f32_to_f16<<<(E_ * 3 * E_ + 255) / 256, 256, 0, stream>>>(
      qkv_w, qkvwT, E_, 3 * E_);
  transpose_f32_to_f16<<<(E_ * E_ + 255) / 256, 256, 0, stream>>>(
      out_w, outwT, E_, E_);
  transpose_f32_to_f16<<<(E_ * F_ + 255) / 256, 256, 0, stream>>>(
      w1, w1T, E_, F_);
  transpose_f32_to_f16<<<(F_ * E_ + 255) / 256, 256, 0, stream>>>(
      w2, w2T, F_, E_);

  // 2) LN1
  layernorm_f16<<<M_, 256, 0, stream>>>(atom_embed, ln1_g, ln1_b, x16);

  // 3) QKV GEMM: (16384 x 1536) = x16 (16384x512) * qkv_w
  gemm_wmma<false, false, false, true>
      <<<dim3((3 * E_) / 256, M_ / 32), 128, 0, stream>>>(
          x16, qkvwT, qkv_b, nullptr, nullptr, qkv16, M_, 3 * E_, E_);

  // 4) attention
  attn_kernel<<<dim3(N_ / 16, H_, B_), 32, 0, stream>>>(
      qkv16, pair_repr, mask, pair_w, pair_b, attn16);

  // 5) out-proj + residual -> x (f32) into d_out
  gemm_wmma<false, true, true, false>
      <<<dim3(E_ / 256, M_ / 32), 128, 0, stream>>>(
          attn16, outwT, out_b, atom_embed, out, nullptr, M_, E_, E_);

  // 6) LN2 on x
  layernorm_f16<<<M_, 256, 0, stream>>>(out, ln2_g, ln2_b, y16);

  // 7) FFN1 with exact GELU -> h16
  gemm_wmma<true, false, false, true>
      <<<dim3(F_ / 256, M_ / 32), 128, 0, stream>>>(
          y16, w1T, b1, nullptr, nullptr, h16, M_, F_, E_);

  // 8) FFN2 + residual(x) -> final output (overwrites x in d_out)
  gemm_wmma<false, true, true, false>
      <<<dim3(E_ / 256, M_ / 32), 128, 0, stream>>>(
          h16, w2T, b2, out, out, nullptr, M_, E_, F_);

  // 9) second tuple element: pair_repr pass-through
  copy_f32x4<<<((size_t)B_ * N_ * N_ / 4) / 256, 256, 0, stream>>>(
      (const float4*)pair_repr, (float4*)(out + (size_t)M_ * E_));
}
